// SEGNN_25340307046987
// MI455X (gfx1250) — matrix operand order repaired and checked by reading, not compiled, not run on hardware
//
#include <hip/hip_runtime.h>
#include <hip/hip_bf16.h>
#include <math.h>

typedef __bf16 bf16;
typedef __attribute__((ext_vector_type(16))) __bf16 v16bf;
typedef __attribute__((ext_vector_type(8)))  float  v8f;

#define WPB 2          // waves per block (wave32)
#define H0C 32
#define H1C 32
#define GG  128
#define POOLC 64

// Weight descriptor: pointers into d_ws holding bf16 B-fragment layouts
struct TPW {
  const bf16* wss;
  const bf16* wvv;
  const bf16* wsv;
  const bf16* wvs;
  const float* bias;
  int k0c;   // K chunks (32) for scalar input
  int k1c;   // K chunks for vector input
  int oAT;   // 16-wide output tiles of out0+gate
  int o1T;   // 16-wide output tiles of out1
  int out0;  // scalar output channels (pre-gate split)
  int gate;  // gated tp?
};

// A-fragment element position within one 16x32 bf16 chunk (ISA 7.12.2):
// VGPR j (pairs), lanes 0-15: K=2j..2j+1 (j<4), K=16+.. (j>=4); lanes 16-31: +8 / +24
__device__ __forceinline__ int a_idx(int m, int kk) {
  int hf = (kk >> 3) & 1;
  int e  = ((kk >> 4) << 3) | (kk & 7);
  return (((hf << 4) + m) << 4) + e;
}

__device__ __forceinline__ v8f wmma_bf16(v16bf a, v16bf b, v8f c) {
  return __builtin_amdgcn_wmma_f32_16x16x32_bf16(false, a, false, b, (short)0, c,
                                                 false, false);
}

__device__ __forceinline__ float sigmoidf_(float x) { return 1.f / (1.f + __expf(-x)); }

// Wave-level fully-connected O(3) tensor product (lmax=1) on a 16-row tile.
// sA : scalar input A-frags   [k0c chunks * 512] bf16 (raw, no a0 scale)
// vA : vector input A-frags   [3 comps][k1c chunks]*512 bf16 (pre-scaled by a0)
// vdA: sum_c v*a1 A-frags     [k1c chunks]*512 bf16
// Scale constants are folded into the weight fragments.
__device__ void tp_wave(const bf16* sA, const bf16* vA, const bf16* vdA,
                        const float* a0, const float* a1, const TPW& w,
                        float* sOut, float* vOut, float* gateLds)
{
  const int lane  = threadIdx.x & 31;
  const int nn    = lane & 15;
  const int mbase = (lane >> 4) << 3;

  // ---- scalar path: s_all = (sA@Wss')*a0 + vdA@Wvv' + bias ----
  for (int ot = 0; ot < w.oAT; ++ot) {
    v8f acc = {};
    for (int k = 0; k < w.k0c; ++k)
      acc = wmma_bf16(*(const v16bf*)(sA + (k << 9) + (lane << 4)),
                      *(const v16bf*)(w.wss + (((k * w.oAT + ot) << 9) + (lane << 4))),
                      acc);
#pragma unroll
    for (int r = 0; r < 8; ++r) acc[r] *= a0[mbase + r];
    for (int k = 0; k < w.k1c; ++k)
      acc = wmma_bf16(*(const v16bf*)(vdA + (k << 9) + (lane << 4)),
                      *(const v16bf*)(w.wvv + (((k * w.oAT + ot) << 9) + (lane << 4))),
                      acc);
    const int n  = (ot << 4) + nn;
    const float bv = w.bias[n];
#pragma unroll
    for (int r = 0; r < 8; ++r) {
      const int m = mbase + r;
      float val = acc[r] + bv;
      if (!w.gate)            sOut[m * w.out0 + n] = val;                 // raw tp
      else if (n < w.out0)    sOut[m * w.out0 + n] = val * sigmoidf_(val); // SiLU
      else                    gateLds[(m << 5) + (n - w.out0)] = sigmoidf_(val);
    }
  }
  __syncthreads();

  // ---- vector path: v_out[.,o,c] = P[.,o]*a1[.,c] + Q_c[.,o] (Q has a0 folded) ----
  for (int ot = 0; ot < w.o1T; ++ot) {
    v8f accP = {};
    for (int k = 0; k < w.k0c; ++k)
      accP = wmma_bf16(*(const v16bf*)(sA + (k << 9) + (lane << 4)),
                       *(const v16bf*)(w.wsv + (((k * w.o1T + ot) << 9) + (lane << 4))),
                       accP);
    const int n = (ot << 4) + nn;
    for (int c = 0; c < 3; ++c) {
      v8f accQ = {};
      for (int k = 0; k < w.k1c; ++k)
        accQ = wmma_bf16(*(const v16bf*)(vA + (((c * w.k1c + k) << 9) + (lane << 4))),
                         *(const v16bf*)(w.wvs + (((k * w.o1T + ot) << 9) + (lane << 4))),
                         accQ);
#pragma unroll
      for (int r = 0; r < 8; ++r) {
        const int m = mbase + r;
        float val = accP[r] * a1[m * 3 + c] + accQ[r];
        if (w.gate) val *= gateLds[(m << 5) + n];
        vOut[(m * (w.o1T << 4) + n) * 3 + c] = val;
      }
    }
  }
  __syncthreads();
}

// ---- convert fp32 weight [rows x cols] into bf16 B-fragment layout with scale ----
__global__ void prep_w(const float* __restrict__ W, int rows, int cols,
                       int kc, int nt, float scale, bf16* __restrict__ dst) {
  int total = kc * 32 * nt * 16;
  for (int i = blockIdx.x * blockDim.x + threadIdx.x; i < total;
       i += gridDim.x * blockDim.x) {
    int k = i / (nt * 16);
    int n = i % (nt * 16);
    float val = (k < rows && n < cols) ? W[k * cols + n] * scale : 0.f;
    int hf = (k >> 3) & 1, e = (((k >> 4) & 1) << 3) | (k & 7);
    int lane = (hf << 4) | (n & 15);
    dst[(((k >> 5) * nt + (n >> 4)) << 9) + (lane << 4) + e] = (bf16)val;
  }
}

// ---- embedding: s = x*Wss0 + b ; v[.,o,c] = x*Wsv0[o]*na1[c] ----
__global__ void embed_kernel(const float* __restrict__ x, const float* __restrict__ na,
                             const float* __restrict__ wss, const float* __restrict__ wsv,
                             const float* __restrict__ bias,
                             float* __restrict__ s, float* __restrict__ v, int n) {
  int t = blockIdx.x * blockDim.x + threadIdx.x;
  if (t >= n * 32) return;
  int node = t >> 5, o = t & 31;
  float xv = x[node];
  s[t] = xv * wss[o] + bias[o];
  float sv = xv * wsv[o];
#pragma unroll
  for (int c = 0; c < 3; ++c) v[t * 3 + c] = sv * na[node * 4 + 1 + c];
}

// ---- fused per-edge message kernel: gather -> msg1 -> msg2 -> scatter-add ----
__global__ __launch_bounds__(WPB * 32) void conv_edge_kernel(
    const float* __restrict__ s, const float* __restrict__ v,
    const int* __restrict__ eidx, const float* __restrict__ eattr,
    const float* __restrict__ addf, TPW m1, TPW m2,
    float* __restrict__ aggS, float* __restrict__ aggV, int E_)
{
  __shared__ __align__(32) bf16 shA[WPB][3 * 512];
  __shared__ __align__(32) bf16 shV[WPB][6 * 512];
  __shared__ __align__(32) bf16 shVD[WPB][2 * 512];
  __shared__ float shG[WPB][16 * 32];
  __shared__ float shS[WPB][16 * 32];
  __shared__ float shVo[WPB][16 * 32 * 3];
  __shared__ int   shSrc[WPB][16], shDst[WPB][16];
  __shared__ float shA0[WPB][16], shA1[WPB][48], shAF[WPB][16];

  const int wv = threadIdx.x >> 5, lane = threadIdx.x & 31;
  const long base = ((long)blockIdx.x * WPB + wv) * 16;

  if (lane < 16) {
    long e = base + lane;
    bool ok = e < (long)E_;
    long ee = ok ? e : 0;
    shSrc[wv][lane] = ok ? eidx[ee] : -1;
    shDst[wv][lane] = ok ? eidx[(long)E_ + ee] : -1;
    shA0[wv][lane]  = ok ? eattr[ee * 4] : 0.f;
    shAF[wv][lane]  = ok ? addf[ee] : 0.f;
    for (int c = 0; c < 3; ++c)
      shA1[wv][lane * 3 + c] = ok ? eattr[ee * 4 + 1 + c] : 0.f;
  }
  __syncthreads();

  // stage ms = [s[dst] | s[src] | add_f | pad] -> [16 x 96] A-frags
  for (int t = lane; t < 16 * 96; t += 32) {
    int m = t / 96, k = t % 96;
    float val = 0.f;
    if (k < 64) {
      int nd = (k < 32) ? shDst[wv][m] : shSrc[wv][m];
      if (nd >= 0) val = s[nd * 32 + (k & 31)];
    } else if (k == 64) val = shAF[wv][m];
    shA[wv][((k >> 5) << 9) + a_idx(m, k & 31)] = (bf16)val;
  }
  // stage mv = [v[dst] | v[src]] (x a0) and vdot = sum_c mv*a1
  for (int t = lane; t < 16 * 64; t += 32) {
    int m = t / 64, i = t % 64;
    int nd = (i < 32) ? shDst[wv][m] : shSrc[wv][m];
    float a0 = shA0[wv][m], vd = 0.f;
#pragma unroll
    for (int c = 0; c < 3; ++c) {
      float val = (nd >= 0) ? v[(nd * 32 + (i & 31)) * 3 + c] : 0.f;
      shV[wv][((c * 2 + (i >> 5)) << 9) + a_idx(m, i & 31)] = (bf16)(val * a0);
      vd += val * shA1[wv][m * 3 + c];
    }
    shVD[wv][((i >> 5) << 9) + a_idx(m, i & 31)] = (bf16)vd;
  }
  __syncthreads();

  tp_wave(shA[wv], shV[wv], shVD[wv], shA0[wv], shA1[wv], m1,
          shS[wv], shVo[wv], shG[wv]);

  // restage msg2 inputs from msg1 outputs
  for (int t = lane; t < 16 * 32; t += 32) {
    int m = t >> 5, k = t & 31;
    shA[wv][a_idx(m, k)] = (bf16)shS[wv][t];
    float a0 = shA0[wv][m], vd = 0.f;
#pragma unroll
    for (int c = 0; c < 3; ++c) {
      float val = shVo[wv][t * 3 + c];
      shV[wv][(c << 9) + a_idx(m, k)] = (bf16)(val * a0);
      vd += val * shA1[wv][m * 3 + c];
    }
    shVD[wv][a_idx(m, k)] = (bf16)vd;
  }
  __syncthreads();

  tp_wave(shA[wv], shV[wv], shVD[wv], shA0[wv], shA1[wv], m2,
          shS[wv], shVo[wv], shG[wv]);

  // scatter-add into node aggregates
  for (int t = lane; t < 16 * 32; t += 32) {
    int m = t >> 5, o = t & 31;
    int d = shDst[wv][m];
    if (d >= 0) {
      atomicAdd(&aggS[d * 32 + o], shS[wv][t]);
#pragma unroll
      for (int c = 0; c < 3; ++c)
        atomicAdd(&aggV[(d * 32 + o) * 3 + c], shVo[wv][t * 3 + c]);
    }
  }
}

// ---- per-node update: [s|agg_s],[v|agg_v] -> upd1(gate) -> upd2 -> residual ----
__global__ __launch_bounds__(WPB * 32) void node_update_kernel(
    float* __restrict__ s, float* __restrict__ v,
    const float* __restrict__ aggS, const float* __restrict__ aggV,
    const float* __restrict__ nattr, TPW u1, TPW u2, int N_)
{
  __shared__ __align__(32) bf16 shA[WPB][2 * 512];
  __shared__ __align__(32) bf16 shV[WPB][6 * 512];
  __shared__ __align__(32) bf16 shVD[WPB][2 * 512];
  __shared__ float shG[WPB][16 * 32];
  __shared__ float shS[WPB][16 * 32];
  __shared__ float shVo[WPB][16 * 32 * 3];
  __shared__ float shA0[WPB][16], shA1[WPB][48];

  const int wv = threadIdx.x >> 5, lane = threadIdx.x & 31;
  const long base = ((long)blockIdx.x * WPB + wv) * 16;

  if (lane < 16) {
    long nd = base + lane;
    bool ok = nd < (long)N_;
    shA0[wv][lane] = 1.f;
    for (int c = 0; c < 3; ++c)
      shA1[wv][lane * 3 + c] = ok ? nattr[nd * 4 + 1 + c] : 0.f;
  }
  __syncthreads();

  for (int t = lane; t < 16 * 64; t += 32) {
    int m = t / 64, k = t % 64;
    long nd = base + m;
    float val = 0.f;
    if (nd < (long)N_) val = (k < 32) ? s[nd * 32 + k] : aggS[nd * 32 + (k - 32)];
    shA[wv][((k >> 5) << 9) + a_idx(m, k & 31)] = (bf16)val;
  }
  for (int t = lane; t < 16 * 64; t += 32) {
    int m = t / 64, i = t % 64;
    long nd = base + m;
    float vd = 0.f;
#pragma unroll
    for (int c = 0; c < 3; ++c) {
      float val = 0.f;
      if (nd < (long)N_)
        val = (i < 32) ? v[(nd * 32 + i) * 3 + c] : aggV[(nd * 32 + (i - 32)) * 3 + c];
      shV[wv][((c * 2 + (i >> 5)) << 9) + a_idx(m, i & 31)] = (bf16)val; // a0 == 1
      vd += val * shA1[wv][m * 3 + c];
    }
    shVD[wv][((i >> 5) << 9) + a_idx(m, i & 31)] = (bf16)vd;
  }
  __syncthreads();

  tp_wave(shA[wv], shV[wv], shVD[wv], shA0[wv], shA1[wv], u1,
          shS[wv], shVo[wv], shG[wv]);

  for (int t = lane; t < 16 * 32; t += 32) {
    int m = t >> 5, k = t & 31;
    shA[wv][a_idx(m, k)] = (bf16)shS[wv][t];
    float vd = 0.f;
#pragma unroll
    for (int c = 0; c < 3; ++c) {
      float val = shVo[wv][t * 3 + c];
      shV[wv][(c << 9) + a_idx(m, k)] = (bf16)val;
      vd += val * shA1[wv][m * 3 + c];
    }
    shVD[wv][a_idx(m, k)] = (bf16)vd;
  }
  __syncthreads();

  tp_wave(shA[wv], shV[wv], shVD[wv], shA0[wv], shA1[wv], u2,
          shS[wv], shVo[wv], shG[wv]);

  for (int t = lane; t < 16 * 32; t += 32) {
    int m = t >> 5, o = t & 31;
    long nd = base + m;
    if (nd < (long)N_) {
      s[nd * 32 + o] += shS[wv][t];
#pragma unroll
      for (int c = 0; c < 3; ++c)
        v[(nd * 32 + o) * 3 + c] += shVo[wv][t * 3 + c];
    }
  }
}

// ---- readout head: pre1(gate) -> pre2 (scalars only, POOL=64) ----
__global__ __launch_bounds__(WPB * 32) void pre_kernel(
    const float* __restrict__ s, const float* __restrict__ v,
    const float* __restrict__ nattr, TPW p1, TPW p2,
    float* __restrict__ sPool, int N_)
{
  __shared__ __align__(32) bf16 shA[WPB][1 * 512];
  __shared__ __align__(32) bf16 shV[WPB][3 * 512];
  __shared__ __align__(32) bf16 shVD[WPB][1 * 512];
  __shared__ float shG[WPB][16 * 32];
  __shared__ float shS[WPB][16 * 64];
  __shared__ float shVo[WPB][16 * 32 * 3];
  __shared__ float shA0[WPB][16], shA1[WPB][48];

  const int wv = threadIdx.x >> 5, lane = threadIdx.x & 31;
  const long base = ((long)blockIdx.x * WPB + wv) * 16;

  if (lane < 16) {
    long nd = base + lane;
    bool ok = nd < (long)N_;
    shA0[wv][lane] = 1.f;
    for (int c = 0; c < 3; ++c)
      shA1[wv][lane * 3 + c] = ok ? nattr[nd * 4 + 1 + c] : 0.f;
  }
  __syncthreads();

  for (int t = lane; t < 16 * 32; t += 32) {
    int m = t >> 5, k = t & 31;
    long nd = base + m;
    float sv = (nd < (long)N_) ? s[nd * 32 + k] : 0.f;
    shA[wv][a_idx(m, k)] = (bf16)sv;
    float vd = 0.f;
#pragma unroll
    for (int c = 0; c < 3; ++c) {
      float val = (nd < (long)N_) ? v[(nd * 32 + k) * 3 + c] : 0.f;
      shV[wv][(c << 9) + a_idx(m, k)] = (bf16)val;
      vd += val * shA1[wv][m * 3 + c];
    }
    shVD[wv][a_idx(m, k)] = (bf16)vd;
  }
  __syncthreads();

  tp_wave(shA[wv], shV[wv], shVD[wv], shA0[wv], shA1[wv], p1,
          shS[wv], shVo[wv], shG[wv]);

  for (int t = lane; t < 16 * 32; t += 32) {
    int m = t >> 5, k = t & 31;
    shA[wv][a_idx(m, k)] = (bf16)shS[wv][m * 32 + k];
    float vd = 0.f;
#pragma unroll
    for (int c = 0; c < 3; ++c) {
      float val = shVo[wv][t * 3 + c];
      shV[wv][(c << 9) + a_idx(m, k)] = (bf16)val;
      vd += val * shA1[wv][m * 3 + c];
    }
    shVD[wv][a_idx(m, k)] = (bf16)vd;
  }
  __syncthreads();

  tp_wave(shA[wv], shV[wv], shVD[wv], shA0[wv], shA1[wv], p2,
          shS[wv], shVo[wv], shG[wv]);

  for (int t = lane; t < 16 * 64; t += 32) {
    int m = t >> 6, o = t & 63;
    long nd = base + m;
    if (nd < (long)N_) sPool[nd * 64 + o] = shS[wv][m * 64 + o];
  }
}

__global__ void pool_accum(const float* __restrict__ sPool, const int* __restrict__ batch,
                           float* __restrict__ pooled, float* __restrict__ cnt, int n) {
  int t = blockIdx.x * blockDim.x + threadIdx.x;
  if (t >= n * 64) return;
  int node = t >> 6, o = t & 63;
  int g = batch[node];
  atomicAdd(&pooled[g * 64 + o], sPool[t]);
  if (o == 0) atomicAdd(&cnt[g], 1.0f);
}

__global__ void post_kernel(const float* __restrict__ pooled, const float* __restrict__ cnt,
                            const float* __restrict__ W1, const float* __restrict__ b1,
                            const float* __restrict__ W2, const float* __restrict__ b2,
                            float* __restrict__ out) {
  int g = threadIdx.x;
  if (g >= GG) return;
  float c = cnt[g]; c = c > 1.f ? c : 1.f;
  float row[POOLC];
#pragma unroll 8
  for (int o = 0; o < POOLC; ++o) row[o] = pooled[g * POOLC + o] / c;
  float acc = 0.f;
  for (int j = 0; j < POOLC; ++j) {
    float h = 0.f;
#pragma unroll 8
    for (int o = 0; o < POOLC; ++o) h += row[o] * W1[o * POOLC + j];
    h = h * 0.125f + b1[j];        // / sqrt(64)
    h = h * sigmoidf_(h);          // SiLU
    acc += h * W2[j];
  }
  out[g] = acc * 0.125f + b2[0];
}

// ===================== host side =====================
static inline size_t alignUp(size_t x, size_t a) { return (x + a - 1) / a * a; }

static TPW make_tp(char* ws, size_t& off, void* const* d_in, int idx,
                   int in0, int in1, int out0, int g, int out1, int gate,
                   hipStream_t stream) {
  TPW t{};
  t.k0c = (in0 + 31) / 32;
  t.k1c = (in1 + 31) / 32;
  t.oAT = (out0 + g) / 16;
  t.o1T = out1 / 16;
  t.out0 = out0;
  t.gate = gate;
  float two = (in0 > 0 && in1 > 0) ? 0.70710678118f : 1.f;
  float sSS = two / sqrtf((float)in0);
  float sVV = (in1 > 0) ? two / sqrtf(3.f * (float)in1) : 0.f;
  float sSV = two / sqrtf((float)in0);
  float sVS = (in1 > 0) ? two / sqrtf((float)in1) : 0.f;
  auto mk = [&](const float* W, int rows, int cols, int kc, int nt,
                float sc) -> const bf16* {
    size_t n = (size_t)kc * nt * 512;
    bf16* dst = (bf16*)(ws + off);
    off = alignUp(off + n * 2, 256);
    int total = (int)n;
    int grid = (total + 255) / 256;
    prep_w<<<grid, 256, 0, stream>>>(W, rows, cols, kc, nt, sc, dst);
    return dst;
  };
  t.wss = mk((const float*)d_in[idx + 0], in0, out0 + g, t.k0c, t.oAT, sSS);
  t.wvv = (in1 > 0) ? mk((const float*)d_in[idx + 1], in1, out0 + g, t.k1c, t.oAT, sVV)
                    : nullptr;
  t.wsv = (out1 > 0) ? mk((const float*)d_in[idx + 2], in0, out1, t.k0c, t.o1T, sSV)
                     : nullptr;
  t.wvs = (out1 > 0 && in1 > 0)
              ? mk((const float*)d_in[idx + 3], in1, out1, t.k1c, t.o1T, sVS)
              : nullptr;
  t.bias = (const float*)d_in[idx + 4];
  return t;
}

extern "C" void kernel_launch(void* const* d_in, const int* in_sizes, int n_in,
                              void* d_out, int out_size, void* d_ws, size_t ws_size,
                              hipStream_t stream) {
  (void)n_in; (void)out_size; (void)ws_size;
  const int N_ = in_sizes[0];
  const int E_ = in_sizes[1] / 2;

  const float* x     = (const float*)d_in[0];
  const int*   eidx  = (const int*)d_in[1];
  const float* eattr = (const float*)d_in[2];
  const float* nattr = (const float*)d_in[3];
  const int*   batch = (const int*)d_in[4];
  const float* addf  = (const float*)d_in[5];

  // params flattened in insertion order:
  // 6..10: embed (Wss,Wvv,Wsv,Wvs,bias); 11+L*20: conv L {msg1,msg2,upd1,upd2}x5
  // 51: pre1; 56: pre2; 61/62: post1 W,b; 63/64: post2 W,b
  const float* eWss  = (const float*)d_in[6];
  const float* eWsv  = (const float*)d_in[8];
  const float* eBias = (const float*)d_in[10];
  const float* W1 = (const float*)d_in[61];
  const float* b1 = (const float*)d_in[62];
  const float* W2 = (const float*)d_in[63];
  const float* b2 = (const float*)d_in[64];

  char* ws = (char*)d_ws;
  size_t off = 0;
  float* s      = (float*)(ws + off); off += (size_t)N_ * 32 * 4;
  float* v      = (float*)(ws + off); off += (size_t)N_ * 96 * 4;
  float* aggS   = (float*)(ws + off); off += (size_t)N_ * 32 * 4;
  float* aggV   = (float*)(ws + off); off += (size_t)N_ * 96 * 4;
  float* sPool  = (float*)(ws + off); off += (size_t)N_ * 64 * 4;
  float* pooled = (float*)(ws + off); off += (size_t)GG * 64 * 4;
  float* cnt    = (float*)(ws + off); off += (size_t)GG * 4;
  off = alignUp(off, 256);

  // weight fragment prep (runs each call; deterministic & idempotent)
  TPW msg1[2], msg2[2], upd1[2], upd2[2];
  for (int L = 0; L < 2; ++L) {
    int b = 11 + L * 20;
    msg1[L] = make_tp(ws, off, d_in, b + 0,  2*H0C+1, 2*H1C, H0C, H1C, H1C, 1, stream);
    msg2[L] = make_tp(ws, off, d_in, b + 5,  H0C,     H1C,   H0C, H1C, H1C, 1, stream);
    upd1[L] = make_tp(ws, off, d_in, b + 10, 2*H0C,   2*H1C, H0C, H1C, H1C, 1, stream);
    upd2[L] = make_tp(ws, off, d_in, b + 15, H0C,     H1C,   H0C, 0,   H1C, 0, stream);
  }
  TPW pre1 = make_tp(ws, off, d_in, 51, H0C, H1C, H0C,   H1C, H1C, 1, stream);
  TPW pre2 = make_tp(ws, off, d_in, 56, H0C, H1C, POOLC, 0,   0,   0, stream);

  // embedding
  {
    int total = N_ * 32;
    embed_kernel<<<(total + 255) / 256, 256, 0, stream>>>(x, nattr, eWss, eWsv,
                                                          eBias, s, v, N_);
  }

  const int edgeBlocks = (E_ + 16 * WPB - 1) / (16 * WPB);
  const int nodeBlocks = (N_ + 16 * WPB - 1) / (16 * WPB);

  for (int L = 0; L < 2; ++L) {
    hipMemsetAsync(aggS, 0, (size_t)N_ * 32 * 4, stream);
    hipMemsetAsync(aggV, 0, (size_t)N_ * 96 * 4, stream);
    conv_edge_kernel<<<edgeBlocks, WPB * 32, 0, stream>>>(s, v, eidx, eattr, addf,
                                                          msg1[L], msg2[L],
                                                          aggS, aggV, E_);
    node_update_kernel<<<nodeBlocks, WPB * 32, 0, stream>>>(s, v, aggS, aggV, nattr,
                                                            upd1[L], upd2[L], N_);
  }

  pre_kernel<<<nodeBlocks, WPB * 32, 0, stream>>>(s, v, nattr, pre1, pre2, sPool, N_);

  hipMemsetAsync(pooled, 0, (size_t)GG * 64 * 4, stream);
  hipMemsetAsync(cnt, 0, (size_t)GG * 4, stream);
  pool_accum<<<(N_ * 64 + 255) / 256, 256, 0, stream>>>(sPool, batch, pooled, cnt, N_);
  post_kernel<<<1, 128, 0, stream>>>(pooled, cnt, W1, b1, W2, b2, (float*)d_out);
}
